// EfficientTransformerBlock_v2_57543971832384
// MI455X (gfx1250) — matrix operand order repaired and checked by reading, not compiled, not run on hardware
//
#include <hip/hip_runtime.h>
#include <hip/hip_bf16.h>
#include <math.h>

// ---------------------------------------------------------------------------
// Types for CDNA5 WMMA (wave32): v_wmma_f32_16x16x32_bf16
// ---------------------------------------------------------------------------
typedef __attribute__((ext_vector_type(16))) __bf16        bf16x16;
typedef __attribute__((ext_vector_type(8)))  float         f32x8;
typedef __attribute__((ext_vector_type(8)))  unsigned short u16x8;

__device__ __forceinline__ unsigned short f2bf(float f) {
  unsigned int u = __builtin_bit_cast(unsigned int, f);
  u = (u + 0x7FFFu + ((u >> 16) & 1u)) >> 16;   // round-to-nearest-even
  return (unsigned short)u;
}
__device__ __forceinline__ float bf2f(unsigned short h) {
  unsigned int u = ((unsigned int)h) << 16;
  return __builtin_bit_cast(float, u);
}

union ABFrag { u16x8 h[2]; bf16x16 v; };

// ---------------------------------------------------------------------------
// CDNA5 async memory->LDS (ASYNCcnt-tracked; ISA 15.18.3 op 98).
// VDST VGPR holds the workgroup-relative LDS byte offset (= low 32 bits of the
// flat shared-aperture pointer); INST_OFFSET is added to BOTH addresses, so a
// 32-byte copy is two b128 ops sharing one address pair.
// ---------------------------------------------------------------------------
__device__ __forceinline__ void async_copy32B(const void* g, void* l) {
  asm volatile("global_load_async_to_lds_b128 %0, %1, off\n\t"
               "global_load_async_to_lds_b128 %0, %1, off offset:16"
               :
               : "v"((unsigned)(unsigned long long)(uintptr_t)l),
                 "v"((unsigned long long)(uintptr_t)g)
               : "memory");
}
__device__ __forceinline__ void wait_async0() {
  asm volatile("s_wait_asynccnt 0x0" ::: "memory");
}

// ---------------------------------------------------------------------------
// Generic tiled WMMA GEMM:  D[b] = sum_t A_t (MxK) * shift_t(B[b]) (KxN) + bias
// Plain GEMMs (ntaps=1) and 3-tap directional convs (ntaps=3, column shift
// +/- shift_stride with edge masking). Double-buffered LDS; A tiles (and
// transposed-B tiles) are staged with global_load_async_to_lds_b128.
// Block: 256 threads = 8 waves; tile 128x128, k-step 32.
// Wave grid 4x2 -> each wave owns 32 rows x 64 cols = 2x4 16x16 fragments.
// ---------------------------------------------------------------------------
#define BM 128
#define BN 128
#define BK 32
#define LDT 40   // padded LDS row stride (elements) to dodge bank conflicts

struct GemmP {
  const unsigned short* A;      // bf16 [tap][M][K]
  const unsigned short* B;      // bf16 activations
  const float* bias;            // [M] or null
  const float* resid;           // f32, same layout/stride as out, or null
  float* out32;                 // optional f32 output
  unsigned short* out16;        // optional bf16 output
  int M, N, K, ldb;
  long astride_b, bstride_b, ostride_b;
  int ntaps, shift_stride, edge_mode, b_transposed;  // edge: 0=[0,N), 1=row-tile
  long tapstride;
};

__global__ __launch_bounds__(256) void gemm_wmma_kernel(GemmP p) {
  __shared__ unsigned short As[2][BM * LDT];
  __shared__ unsigned short Bs[2][BN * LDT];

  const int tid  = threadIdx.x;
  const int b    = blockIdx.z;
  const int n0   = blockIdx.x * BN;
  const int m0   = blockIdx.y * BM;
  const int lane = tid & 31;
  const int wave = tid >> 5;
  const int wm   = wave & 3;    // 4 row-groups of 32
  const int wn   = wave >> 2;   // 2 col-groups of 64
  const int half = lane >> 4;   // K/M-half select per ISA 7.12.2 layout
  const int r    = lane & 15;

  f32x8 acc[2][4];
#pragma unroll
  for (int i = 0; i < 2; ++i)
#pragma unroll
    for (int j = 0; j < 4; ++j)
#pragma unroll
      for (int e = 0; e < 8; ++e) acc[i][j][e] = 0.0f;

  const unsigned short* Abase = p.A + (long)b * p.astride_b;
  const unsigned short* Bb    = p.B + (long)b * p.bstride_b;
  const int halfT  = p.ntaps >> 1;
  const int ntiles = p.ntaps * (p.K / BK);

  // ---- stage k-tile (tap t, k-offset k0) into LDS buffer `buf` ----
  auto stage = [&](int t, int k0, int buf) {
    const unsigned short* At = Abase + (long)t * p.tapstride;
    // A tile: async 32B per thread (row-major contiguous)
    {
      const int row = tid >> 1, seg = tid & 1;
      const unsigned short* src = At + (long)(m0 + row) * p.K + k0 + seg * 16;
      async_copy32B(src, &As[buf][row * LDT + seg * 16]);
    }
    if (p.b_transposed) {
      // B given as [Ncols][K]: pure block copy -> async too
      const int nrow = tid >> 1, seg = tid & 1;
      const unsigned short* src = Bb + (long)(n0 + nrow) * p.ldb + k0 + seg * 16;
      async_copy32B(src, &Bs[buf][nrow * LDT + seg * 16]);
    } else {
      // B is [K][ldb] row-major: transpose into N-major LDS with optional
      // conv-tap shift + edge masking. Vector global path when the shift keeps
      // 16B alignment and the 16-column run is fully interior.
      const int shift = (t - halfT) * p.shift_stride;
      const int glo = (p.edge_mode == 1) ? n0 : 0;
      const int ghi = (p.edge_mode == 1) ? (n0 + BN) : p.N;
      const int krow = tid >> 3;   // 0..31
      const int seg  = tid & 7;    // 8 segs * 16 cols
      const unsigned short* srow = Bb + (long)(k0 + krow) * p.ldb;
      if (k0 + BK < p.K) __builtin_prefetch(srow + (long)BK * p.ldb, 0, 1);
      const int gbase = n0 + seg * 16 + shift;
      unsigned short tmp[16];
      if (((shift & 7) == 0) && gbase >= glo && (gbase + 16) <= ghi) {
        const uint4* s4 = (const uint4*)(srow + gbase);
        *(uint4*)&tmp[0] = s4[0];
        *(uint4*)&tmp[8] = s4[1];
      } else {
#pragma unroll
        for (int j = 0; j < 16; ++j) {
          const int g = gbase + j;
          tmp[j] = (g >= glo && g < ghi) ? srow[g] : (unsigned short)0;
        }
      }
#pragma unroll
      for (int j = 0; j < 16; ++j)
        Bs[buf][(seg * 16 + j) * LDT + krow] = tmp[j];
    }
  };

  stage(0, 0, 0);                 // prologue: tile 0 in flight
  int ts = 0, ks = BK;            // coords of next tile to stage (incremental,
  if (ks == p.K) { ks = 0; ++ts; } // no division on the critical path)

  for (int it = 0; it < ntiles; ++it) {
    const int buf = it & 1;
    wait_async0();        // this wave's async writes into buf have landed
    __syncthreads();      // everyone's staging of tile `it` is visible
    if (it + 1 < ntiles) {            // DMA next tile during the WMMAs
      stage(ts, ks, buf ^ 1);
      ks += BK;
      if (ks == p.K) { ks = 0; ++ts; }
    }

    // ---- fragment loads per ISA 7.12.2 (wave32, 16-bit A 16x32 / B 32x16)
    const unsigned short* asb = &As[buf][0];
    const unsigned short* bsb = &Bs[buf][0];
    ABFrag a0, a1, bf[4];
    {
      const unsigned short* ap0 = asb + (wm * 32 + r) * LDT + 8 * half;
      a0.h[0] = *(const u16x8*)ap0;
      a0.h[1] = *(const u16x8*)(ap0 + 16);
      const unsigned short* ap1 = asb + (wm * 32 + 16 + r) * LDT + 8 * half;
      a1.h[0] = *(const u16x8*)ap1;
      a1.h[1] = *(const u16x8*)(ap1 + 16);
    }
#pragma unroll
    for (int nf = 0; nf < 4; ++nf) {
      const unsigned short* bp = bsb + (wn * 64 + nf * 16 + r) * LDT + 8 * half;
      bf[nf].h[0] = *(const u16x8*)bp;
      bf[nf].h[1] = *(const u16x8*)(bp + 16);
    }
#pragma unroll
    for (int nf = 0; nf < 4; ++nf) {
      acc[0][nf] = __builtin_amdgcn_wmma_f32_16x16x32_bf16(
          false, a0.v, false, bf[nf].v, (short)0, acc[0][nf], false, false);
      acc[1][nf] = __builtin_amdgcn_wmma_f32_16x16x32_bf16(
          false, a1.v, false, bf[nf].v, (short)0, acc[1][nf], false, false);
    }
    __syncthreads();      // all waves consumed buf before it is re-staged
  }

  // ---- epilogue: C/D layout -> row = v + 8*half, col = lane&15 ----
#pragma unroll
  for (int mf = 0; mf < 2; ++mf)
#pragma unroll
    for (int nf = 0; nf < 4; ++nf)
#pragma unroll
      for (int v = 0; v < 8; ++v) {
        const int row = m0 + wm * 32 + mf * 16 + v + 8 * half;
        const int col = n0 + wn * 64 + nf * 16 + r;
        const long idx = (long)b * p.ostride_b + (long)row * p.N + col;
        float x = acc[mf][nf][v];
        if (p.bias)  x += p.bias[row];
        if (p.resid) x += p.resid[idx];
        if (p.out32) p.out32[idx] = x;
        if (p.out16) p.out16[idx] = f2bf(x);
      }
}

// ---------------------------------------------------------------------------
// Elementwise / reduction kernels
// ---------------------------------------------------------------------------
__global__ void pack_bf16_kernel(const float* src, unsigned short* dst, long n) {
  long i = (long)blockIdx.x * blockDim.x + threadIdx.x;
  if (i < n) dst[i] = f2bf(src[i]);
}

// Pack K/Q conv weights tap-major: dst[t][m][c]; m<256 -> K-conv, else Q-conv.
// Source layouts (KC,C,1,3) and (KC,C,3,1) share flat index (m*C+c)*3+t.
__global__ void pack_kq_kernel(const float* wk, const float* wq,
                               const float* bk, const float* bq,
                               unsigned short* dstA, float* dstBias, int C) {
  long i = (long)blockIdx.x * blockDim.x + threadIdx.x;
  const long total = 3L * 512 * C;
  if (i < total) {
    const int t = (int)(i / (512L * C));
    const int rem = (int)(i % (512L * C));
    const int m = rem / C, c = rem % C;
    const float* src = (m < 256) ? wk : wq;
    const int mm = (m < 256) ? m : (m - 256);
    dstA[i] = f2bf(src[(mm * C + c) * 3 + t]);
  }
  if (i < 512) dstBias[i] = (i < 256) ? bk[i] : bq[i - 256];
}

// LayerNorm over C at each (b, n); input f32 (b,C,N), output bf16 (b,C,N).
__global__ void ln_kernel(const float* x, const float* g, const float* beta,
                          unsigned short* out, int C, int N) {
  const long t = (long)blockIdx.x * blockDim.x + threadIdx.x;  // over B*N
  const long b = t / N;
  const int n = (int)(t % N);
  const float* p = x + b * (long)C * N + n;
  float s = 0.f, ss = 0.f;
  for (int c = 0; c < C; ++c) { float v = p[(long)c * N]; s += v; ss += v * v; }
  const float m = s / C;
  const float inv = rsqrtf(fmaxf(ss / C - m * m, 0.f) + 1e-5f);
  unsigned short* o = out + b * (long)C * N + n;
  for (int c = 0; c < C; ++c)
    o[(long)c * N] = f2bf((p[(long)c * N] - m) * inv * g[c] + beta[c]);
}

// Softmax over the spatial axis (row of length N); one block per (b, kc).
__global__ void softmax_row_kernel(const float* in, unsigned short* out, int N,
                                   long in_bstride, long out_bstride, int rows) {
  const int b = blockIdx.x / rows, rr = blockIdx.x % rows;
  const float* rp = in + b * in_bstride + (long)rr * N;
  unsigned short* op = out + b * out_bstride + (long)rr * N;
  __shared__ float sred[256];
  const int tid = threadIdx.x;
  float m = -3.4e38f;
  for (int i = tid; i < N; i += 256) m = fmaxf(m, rp[i]);
  sred[tid] = m; __syncthreads();
  for (int s = 128; s > 0; s >>= 1) {
    if (tid < s) sred[tid] = fmaxf(sred[tid], sred[tid + s]);
    __syncthreads();
  }
  const float mx = sred[0]; __syncthreads();
  float sum = 0.f;
  for (int i = tid; i < N; i += 256) sum += __expf(rp[i] - mx);
  sred[tid] = sum; __syncthreads();
  for (int s = 128; s > 0; s >>= 1) {
    if (tid < s) sred[tid] += sred[tid + s];
    __syncthreads();
  }
  const float inv = 1.f / sred[0];
  for (int i = tid; i < N; i += 256) op[i] = f2bf(__expf(rp[i] - mx) * inv);
}

// Softmax over the channel axis at each (b, n).
__global__ void softmax_col_kernel(const float* in, unsigned short* out, int C,
                                   int N, long in_bstride, long out_bstride) {
  const long t = (long)blockIdx.x * blockDim.x + threadIdx.x;
  const long b = t / N;
  const int n = (int)(t % N);
  const float* p = in + b * in_bstride + n;
  float mx = -3.4e38f;
  for (int c = 0; c < C; ++c) mx = fmaxf(mx, p[(long)c * N]);
  float sum = 0.f;
  for (int c = 0; c < C; ++c) sum += __expf(p[(long)c * N] - mx);
  const float inv = 1.f / sum;
  unsigned short* o = out + b * out_bstride + n;
  for (int c = 0; c < C; ++c)
    o[(long)c * N] = f2bf(__expf(p[(long)c * N] - mx) * inv);
}

// amap[b][n] (set or +=) = scale * sum_kc key[b][kc][n]
__global__ void amap_kernel(const unsigned short* key, float* amap, int KC,
                            int N, long kbstride, float scale, int add) {
  const long t = (long)blockIdx.x * blockDim.x + threadIdx.x;
  const long b = t / N;
  const int n = (int)(t % N);
  const unsigned short* p = key + b * kbstride + n;
  float s = 0.f;
  for (int c = 0; c < KC; ++c) s += bf2f(p[(long)c * N]);
  float* o = amap + b * (long)N + n;
  const float v = s * scale;
  *o = add ? (*o + v) : v;
}

// ctxT[b][v][k] = bf16(ctx[b][k][v])
__global__ void ctxT_kernel(const float* ctx, unsigned short* ctxT, int KC, int VC) {
  const long i = (long)blockIdx.x * blockDim.x + threadIdx.x;
  const long b = i / ((long)KC * VC);
  const int rem = (int)(i % ((long)KC * VC));
  const int v = rem / KC, k = rem % KC;
  ctxT[i] = f2bf(ctx[b * (long)KC * VC + (long)k * VC + v]);
}

// Depthwise 3x3 (pad 1) + exact GELU; bf16 in/out.
__global__ void dw_gelu_kernel(const unsigned short* h1, const float* wdw,
                               const float* bdw, unsigned short* act, int HID,
                               int H, int W) {
  const long t = (long)blockIdx.x * blockDim.x + threadIdx.x;
  const int N = H * W;
  const int n = (int)(t % N);
  const long cb = t / N;
  const int ch = (int)(cb % HID);
  const int hi = n / W, wi = n % W;
  const unsigned short* base = h1 + (t - n);   // start of this (b,ch) plane
  const float* wk = wdw + ch * 9;
  float acc2 = bdw[ch];
#pragma unroll
  for (int dy = -1; dy <= 1; ++dy)
#pragma unroll
    for (int dx = -1; dx <= 1; ++dx) {
      const int y = hi + dy, x = wi + dx;
      if (y >= 0 && y < H && x >= 0 && x < W)
        acc2 += bf2f(base[y * W + x]) * wk[(dy + 1) * 3 + (dx + 1)];
    }
  act[t] = f2bf(0.5f * acc2 * (1.f + erff(acc2 * 0.70710678118654752f)));
}

// ---------------------------------------------------------------------------
// Host orchestration
// ---------------------------------------------------------------------------
extern "C" void kernel_launch(void* const* d_in, const int* in_sizes, int n_in,
                              void* d_out, int out_size, void* d_ws,
                              size_t ws_size, hipStream_t stream) {
  (void)in_sizes; (void)n_in; (void)out_size; (void)ws_size;
  const int Bn = 8, C = 256, H = 128, W = 128, N = H * W;
  const int KC = 256, VC = 256, HID = 1024;

  const float* x    = (const float*)d_in[0];
  const float* wk1  = (const float*)d_in[3];
  const float* bk1  = (const float*)d_in[4];
  const float* wk2  = (const float*)d_in[5];
  const float* bk2  = (const float*)d_in[6];
  const float* wq1  = (const float*)d_in[7];
  const float* bq1  = (const float*)d_in[8];
  const float* wq2  = (const float*)d_in[9];
  const float* bq2  = (const float*)d_in[10];
  const float* wv   = (const float*)d_in[11];
  const float* bv   = (const float*)d_in[12];
  const float* wr   = (const float*)d_in[13];
  const float* brp  = (const float*)d_in[14];
  const float* g1   = (const float*)d_in[15];
  const float* b1   = (const float*)d_in[16];
  const float* g2   = (const float*)d_in[17];
  const float* b2   = (const float*)d_in[18];
  const float* wfc1 = (const float*)d_in[19];
  const float* bfc1 = (const float*)d_in[20];
  const float* wdw  = (const float*)d_in[21];
  const float* bdw  = (const float*)d_in[22];
  const float* wfc2 = (const float*)d_in[23];
  const float* bfc2 = (const float*)d_in[24];

  // ---- workspace arena (phase-aliased) ----
  char* wsp = (char*)d_ws;
  auto arena = [&](size_t bytes) {
    char* p = wsp;
    wsp += (bytes + 255) & ~(size_t)255;
    return p;
  };
  unsigned short* xn   = (unsigned short*)arena((size_t)Bn * C * N * 2);
  unsigned short* val  = (unsigned short*)arena((size_t)Bn * VC * N * 2);
  float*          kq   = (float*)arena((size_t)Bn * 2 * KC * N * 4);
  unsigned short* key  = (unsigned short*)arena((size_t)Bn * KC * N * 2);
  unsigned short* qry  = (unsigned short*)arena((size_t)Bn * KC * N * 2);
  unsigned short* av   = (unsigned short*)arena((size_t)Bn * 2 * VC * N * 2);
  float*          tx   = (float*)arena((size_t)Bn * C * N * 4);
  float*          ctxf = (float*)arena((size_t)Bn * KC * VC * 4);
  unsigned short* ctxT = (unsigned short*)arena((size_t)Bn * VC * KC * 2);
  unsigned short* wkqp = (unsigned short*)arena((size_t)2 * 3 * 512 * C * 2);
  float*          kqb  = (float*)arena((size_t)2 * 512 * 4);
  unsigned short* wvp  = (unsigned short*)arena((size_t)VC * C * 2);
  unsigned short* wrp  = (unsigned short*)arena((size_t)C * 2 * VC * 2);
  unsigned short* wf1p = (unsigned short*)arena((size_t)HID * C * 2);
  unsigned short* wf2p = (unsigned short*)arena((size_t)C * HID * 2);
  // FFN-phase aliases over dead attention buffers:
  unsigned short* n2  = xn;                    // 67 MB
  unsigned short* h1  = (unsigned short*)kq;   // 268 MB
  unsigned short* act = key;                   // key+qry+av = 268 MB contiguous

  float* out_mx   = (float*)d_out;
  float* out_ctx  = out_mx + (size_t)Bn * C * N;
  float* out_amap = out_ctx + (size_t)Bn * KC * VC;

  // ---- weight packing (f32 -> bf16) ----
  {
    long n;
    n = (long)VC * C;
    pack_bf16_kernel<<<(n + 255) / 256, 256, 0, stream>>>(wv, wvp, n);
    n = (long)C * 2 * VC;
    pack_bf16_kernel<<<(n + 255) / 256, 256, 0, stream>>>(wr, wrp, n);
    n = (long)HID * C;
    pack_bf16_kernel<<<(n + 255) / 256, 256, 0, stream>>>(wfc1, wf1p, n);
    n = (long)C * HID;
    pack_bf16_kernel<<<(n + 255) / 256, 256, 0, stream>>>(wfc2, wf2p, n);
    const long nk = 3L * 512 * C;
    pack_kq_kernel<<<(nk + 255) / 256, 256, 0, stream>>>(wk1, wq1, bk1, bq1,
                                                         wkqp, kqb, C);
    pack_kq_kernel<<<(nk + 255) / 256, 256, 0, stream>>>(wk2, wq2, bk2, bq2,
                                                         wkqp + nk, kqb + 512, C);
  }

  auto gemm = [&](const GemmP& p) {
    dim3 grid(p.N / BN, p.M / BM, Bn);
    gemm_wmma_kernel<<<grid, 256, 0, stream>>>(p);
  };

  // LN1: x -> xn (bf16)
  ln_kernel<<<(Bn * N) / 256, 256, 0, stream>>>(x, g1, b1, xn, C, N);

  // V 1x1: val = wv*xn + bv (bf16)
  {
    GemmP p{}; p.A = wvp; p.B = xn; p.bias = bv; p.out16 = val;
    p.M = VC; p.N = N; p.K = C; p.ldb = N;
    p.bstride_b = (long)C * N; p.ostride_b = (long)VC * N; p.ntaps = 1;
    gemm(p);
  }

  for (int brc = 0; brc < 2; ++brc) {
    // K/Q directional conv (3 shifted GEMMs), K rows then Q rows -> kq f32
    {
      GemmP p{}; p.A = wkqp + (long)brc * 3 * 512 * C; p.B = xn;
      p.bias = kqb + brc * 512; p.out32 = kq;
      p.M = 512; p.N = N; p.K = C; p.ldb = N;
      p.bstride_b = (long)C * N; p.ostride_b = (long)512 * N;
      p.ntaps = 3; p.tapstride = (long)512 * C;
      p.shift_stride = (brc == 0) ? 1 : W;
      p.edge_mode = (brc == 0) ? 1 : 0;   // row-edge vs global-edge
      gemm(p);
    }
    // key softmax over spatial axis -> key (bf16)
    softmax_row_kernel<<<Bn * KC, 256, 0, stream>>>(
        kq, key, N, (long)512 * N, (long)KC * N, KC);
    // amap accumulation (branch0 sets, branch1 adds)
    amap_kernel<<<(Bn * N) / 256, 256, 0, stream>>>(
        key, out_amap, KC, N, (long)KC * N, 1.f / (2.f * KC), brc);
    // query softmax over channel axis -> qry (bf16)
    softmax_col_kernel<<<(Bn * N) / 256, 256, 0, stream>>>(
        kq + (long)KC * N, qry, KC, N, (long)512 * N, (long)KC * N);
    // ctx = key (KCxN) * val^T (NxVC): B given transposed (val is VCxN)
    {
      GemmP p{}; p.A = key; p.B = val; p.out32 = ctxf;
      p.M = KC; p.N = VC; p.K = N; p.ldb = N;
      p.astride_b = (long)KC * N; p.bstride_b = (long)VC * N;
      p.ostride_b = (long)KC * VC; p.ntaps = 1; p.b_transposed = 1;
      gemm(p);
    }
    if (brc == 1)  // reference returns the last branch's ctx
      hipMemcpyAsync(out_ctx, ctxf, (size_t)Bn * KC * VC * sizeof(float),
                     hipMemcpyDeviceToDevice, stream);
    // ctx^T pack to bf16
    ctxT_kernel<<<((long)Bn * KC * VC) / 256, 256, 0, stream>>>(ctxf, ctxT, KC, VC);
    // av = ctx^T (VCxKC) * qry (KCxN) -> av rows [brc*VC, brc*VC+VC)
    {
      GemmP p{}; p.A = ctxT; p.B = qry; p.out16 = av + (long)brc * VC * N;
      p.M = VC; p.N = N; p.K = KC; p.ldb = N;
      p.astride_b = (long)VC * KC; p.bstride_b = (long)KC * N;
      p.ostride_b = (long)2 * VC * N; p.ntaps = 1;
      gemm(p);
    }
  }

  // attn = wr * agg + br ; tx = x + attn (f32)
  {
    GemmP p{}; p.A = wrp; p.B = av; p.bias = brp; p.resid = x; p.out32 = tx;
    p.M = C; p.N = N; p.K = 2 * VC; p.ldb = N;
    p.bstride_b = (long)2 * VC * N; p.ostride_b = (long)C * N; p.ntaps = 1;
    gemm(p);
  }

  // LN2: tx -> n2 (bf16, aliases xn)
  ln_kernel<<<(Bn * N) / 256, 256, 0, stream>>>(tx, g2, b2, n2, C, N);

  // fc1: h1 = wfc1 * n2 + bfc1 (bf16, aliases kq)
  {
    GemmP p{}; p.A = wf1p; p.B = n2; p.bias = bfc1; p.out16 = h1;
    p.M = HID; p.N = N; p.K = C; p.ldb = N;
    p.bstride_b = (long)C * N; p.ostride_b = (long)HID * N; p.ntaps = 1;
    gemm(p);
  }

  // depthwise 3x3 + exact GELU -> act (bf16)
  dw_gelu_kernel<<<((long)Bn * HID * N) / 256, 256, 0, stream>>>(
      h1, wdw, bdw, act, HID, H, W);

  // fc2 + residual: mx = tx + wfc2*act + bfc2 -> d_out (f32)
  {
    GemmP p{}; p.A = wf2p; p.B = act; p.bias = bfc2; p.resid = tx;
    p.out32 = out_mx;
    p.M = C; p.N = N; p.K = HID; p.ldb = N;
    p.bstride_b = (long)HID * N; p.ostride_b = (long)C * N; p.ntaps = 1;
    gemm(p);
  }
}
